// Seq2Seq_61357902790729
// MI455X (gfx1250) — matrix operand order repaired and checked by reading, not compiled, not run on hardware
//
#include <hip/hip_runtime.h>
#include <hip/hip_bf16.h>
#include <stdint.h>

#define SOS 0
#define HIDDEN 512
#define EMBED 256
#define V_TGT 50257
#define S_LEN 512
#define T_LEN 512
#define NWG 32          // persistent recurrence workgroups

typedef __bf16 bf16;
typedef bf16 v16bf __attribute__((ext_vector_type(16)));
typedef float v8f   __attribute__((ext_vector_type(8)));
typedef unsigned int v4u __attribute__((ext_vector_type(4)));
typedef int v4i __attribute__((ext_vector_type(4)));
typedef int v8i __attribute__((ext_vector_type(8)));

union Frag { v16bf v; v4u q[2]; };

#if __has_builtin(__builtin_amdgcn_tensor_load_to_lds) && __has_builtin(__builtin_amdgcn_s_wait_tensorcnt)
#define HAVE_TDM 1
#else
#define HAVE_TDM 0
#endif

static __device__ __forceinline__ bf16 f2bf(float f) {
  union { float f; unsigned u; } a; a.f = f;
  unsigned r = a.u + 0x7FFFu + ((a.u >> 16) & 1u);   // RNE
  union { unsigned short s; bf16 b; } o; o.s = (unsigned short)(r >> 16);
  return o.b;
}
static __device__ __forceinline__ float bf2f(bf16 x) {
  union { bf16 b; unsigned short s; } u; u.b = x;
  union { unsigned u32; float f; } r; r.u32 = ((unsigned)u.s) << 16;
  return r.f;
}
static __device__ __forceinline__ float sigm(float x) { return 1.f / (1.f + expf(-x)); }

#if HAVE_TDM
// Issue one 2D TDM load: tile (64 rows x K cols) of bf16 at `gaddr`
// (row stride K elements) into LDS offset lds_off. tensor_dim1=rows_avail
// so rows past the end of A are zero-filled by the TDM's OOB handling.
// This toolchain's builtin is the 6-arg form:
//   (uint32x4 g0, int32x8 g1, int32x4 g2, int32x4 g3, int32x8 pad, i32 cpol)
static __device__ __forceinline__ void tdm_load_2d(uint32_t lds_off,
                                                   const void* gaddr,
                                                   int rows_avail, int K) {
  uint64_t ga = (uint64_t)(uintptr_t)gaddr;
  v4u g0;
  g0.x = 1u;                                        // count=1, user D#
  g0.y = lds_off;                                   // lds_addr (bytes)
  g0.z = (uint32_t)ga;                              // global_addr[31:0]
  g0.w = (uint32_t)((ga >> 32) & 0x01FFFFFFu) | (2u << 30);  // addr[56:32] | type=2
  v8i g1;
  g1[0] = 0x00010000;                               // wg_mask=0, data_size=1 (2 bytes)
  g1[1] = (int)((unsigned)(K & 0xFFFF) << 16);      // tensor_dim0[15:0] @ [31:16]
  g1[2] = (int)(((unsigned)(K >> 16) & 0xFFFFu) |   // tensor_dim0[31:16]
                ((unsigned)(rows_avail & 0xFFFF) << 16));     // tensor_dim1[15:0]
  g1[3] = (int)(((unsigned)(rows_avail >> 16) & 0xFFFFu) |    // tensor_dim1[31:16]
                ((unsigned)(K & 0xFFFF) << 16));    // tile_dim0 = K
  g1[4] = 64;                                       // tile_dim1 = 64, tile_dim2 = 0
  g1[5] = K;                                        // tensor_dim0_stride[31:0]
  g1[6] = 0;                                        // stride hi, dim1_stride lo
  g1[7] = 0;
  v4i z4 = {0, 0, 0, 0};                            // groups 2/3 unused (2D)
  v8i z8 = {0, 0, 0, 0, 0, 0, 0, 0};
  __builtin_amdgcn_tensor_load_to_lds(g0, g1, z4, z4, z8, 0);
}
#endif

// ---------------- init: zero the device-wide barrier ----------------
__global__ void k_init(uint32_t* bar) {
  if (threadIdx.x < 2) bar[threadIdx.x] = 0u;
}

// ---------------- encoder: one GRU step with h0 == 0 ----------------
__global__ void __launch_bounds__(512) k_encoder(
    const int* __restrict__ source, const float* __restrict__ emb_e,
    const float* __restrict__ Wih_e, const float* __restrict__ bih_e,
    const float* __restrict__ bhh_e, float* __restrict__ h_glob) {
  __shared__ float sx[EMBED];
  int tid = threadIdx.x;
  int tok = source[S_LEN - 1];
  if (tid < EMBED) sx[tid] = emb_e[(size_t)tok * EMBED + tid];
  __syncthreads();
  int j = tid;
  float g[3];
#pragma unroll
  for (int gate = 0; gate < 3; ++gate) {
    const float* w = Wih_e + (size_t)(gate * HIDDEN + j) * EMBED;
    float acc = bih_e[gate * HIDDEN + j];
    for (int c = 0; c < EMBED; ++c) acc += w[c] * sx[c];
    g[gate] = acc;
  }
  float r = sigm(g[0] + bhh_e[j]);
  float z = sigm(g[1] + bhh_e[HIDDEN + j]);
  float n = tanhf(g[2] + r * bhh_e[2 * HIDDEN + j]);
  h_glob[j] = (1.f - z) * n;   // + z*h0, h0 == 0
}

// ------------- decoder inputs: X[t] = relu(emb_d[tok_t]) as bf16 -------------
__global__ void k_embed(const int* __restrict__ target,
                        const float* __restrict__ emb_d, bf16* __restrict__ Xbf) {
  int t = blockIdx.x, e = threadIdx.x;
  int tok = (t == 0) ? SOS : target[t - 1];
  float v = emb_d[(size_t)tok * EMBED + e];
  Xbf[(size_t)t * EMBED + e] = f2bf(v > 0.f ? v : 0.f);
}

// ---------------- generic f32 -> bf16 conversion ----------------
__global__ void k_f2bf(const float* __restrict__ in, bf16* __restrict__ out, int n) {
  int i = blockIdx.x * blockDim.x + threadIdx.x;
  if (i < n) out[i] = f2bf(in[i]);
}

// ---------------- block-cooperative WMMA bf16 GEMM, transposed store ----------
// D[n*M + m] = sum_k A[m*K + k] * Bt[n*K + k] + bias[m]
// Block = 8 waves. Block owns a 64-row M stripe (TDM-staged into LDS once),
// each wave computes a 64(M) x 16(N) tile; B streams from global, double-
// buffered in registers. Grid = (M/64 stripes) x (N/128 stripes).
__global__ void __launch_bounds__(256) k_wmma_gemm(
    const bf16* __restrict__ A, const bf16* __restrict__ Bt,
    const float* __restrict__ bias, float* __restrict__ D,
    int M, int N, int K) {
  __shared__ bf16 s_A[64 * 512];   // 64 KB; GEMM1 uses 64*256 of it

  const int NS = N >> 7;                    // N stripes of 128
  int ms = blockIdx.x / NS;
  int ns = blockIdx.x % NS;
  int tid = threadIdx.x;
  int wave = tid >> 5;
  int lane = tid & 31;
  int half = lane >> 4, lrow = lane & 15;

  // ---- stage A stripe [ms*64 .. ms*64+64) x K into LDS ----
#if HAVE_TDM
  if (tid == 0) {
    tdm_load_2d((uint32_t)(uintptr_t)(void*)s_A,
                A + (size_t)ms * 64 * K, M - ms * 64, K);
  }
  __builtin_amdgcn_s_wait_tensorcnt(0);     // trivial for waves 1..7
#else
  for (int idx = tid; idx < 64 * K; idx += 256) {
    int r = idx / K, c = idx - r * K;
    int gr = ms * 64 + r;
    s_A[idx] = (gr < M) ? A[(size_t)gr * K + c] : f2bf(0.f);
  }
#endif
  __syncthreads();

  // ---- per-wave 64x16 tile ----
  int nt = ns * 8 + wave;                   // global N tile
  const bf16* pB = Bt + (size_t)(nt * 16 + lrow) * K + half * 16;

  v8f acc[4] = {{0,0,0,0,0,0,0,0},{0,0,0,0,0,0,0,0},
                {0,0,0,0,0,0,0,0},{0,0,0,0,0,0,0,0}};

  Frag b0, b1;
  b0.q[0] = *(const v4u*)(pB);
  b0.q[1] = *(const v4u*)(pB + 8);

  for (int k0 = 0; k0 < K; k0 += 32) {
    if (k0 + 32 < K) {                      // prefetch next B frag (registers)
      b1.q[0] = *(const v4u*)(pB + k0 + 32);
      b1.q[1] = *(const v4u*)(pB + k0 + 40);
      __builtin_prefetch((const void*)(pB + k0 + 160), 0, 3);
    }
#pragma unroll
    for (int s = 0; s < 4; ++s) {
      const bf16* pa = s_A + (size_t)(s * 16 + lrow) * K + k0 + half * 16;
      Frag a;
      a.q[0] = *(const v4u*)(pa);           // ds_load_b128
      a.q[1] = *(const v4u*)(pa + 8);
      acc[s] = __builtin_amdgcn_wmma_f32_16x16x32_bf16(
          false, a.v, false, b0.v, (short)0, acc[s], false, false);
    }
    b0 = b1;
  }

  // ---- store: lane<16 -> (M=mb+v, N=nt*16+lane); lane>=16 -> M += 8 ----
  int n = nt * 16 + lrow;
  float* outr = D + (size_t)n * M;
#pragma unroll
  for (int s = 0; s < 4; ++s) {
    int mb = ms * 64 + s * 16 + half * 8;
#pragma unroll
    for (int v = 0; v < 8; ++v) {
      int m = mb + v;
      if (m < M) outr[m] = acc[s][v] + bias[m];
    }
  }
}

// ---------------- persistent GRU recurrence ----------------
__global__ void __launch_bounds__(192) k_recurrence(
    const bf16* __restrict__ WhhBf, const float* __restrict__ bhh,
    const float* __restrict__ GiT, float* h_glob,
    bf16* __restrict__ Hbf, uint32_t* bar) {
  __shared__ bf16 s_whh[48 * 512];   // 48 KB
  __shared__ float s_h[512];
  __shared__ float s_gh[48];
  int tid = threadIdx.x;
  int k = blockIdx.x;
  int jbase = k * 16;

  for (int idx = tid; idx < 48 * 512; idx += 192) {
    int l = idx >> 9, c = idx & 511;
    int gate = l >> 4, jl = l & 15;
    s_whh[idx] = WhhBf[(size_t)(gate * HIDDEN + jbase + jl) * HIDDEN + c];
  }
  for (int i = tid; i < 512; i += 192) s_h[i] = h_glob[i];
  __syncthreads();

  int l = tid >> 2, seg = tid & 3;          // 48 rows x 4 K-segments
  int rot = (tid & 31) << 2;                // stagger LDS banks

  for (int t = 0; t < T_LEN; ++t) {
    float acc = 0.f;
    const int base = l * 512 + seg * 128;
#pragma unroll 8
    for (int i = 0; i < 128; ++i) {
      int c = (i + rot) & 127;
      acc += bf2f(s_whh[base + c]) * s_h[seg * 128 + c];
    }
    acc += __shfl_xor(acc, 1, 32);
    acc += __shfl_xor(acc, 2, 32);
    if (seg == 0) s_gh[l] = acc;
    __syncthreads();

    if (tid < 16) {
      int j = jbase + tid;
      const float* git = GiT + (size_t)t * (3 * HIDDEN);
      float ghr = s_gh[tid]      + bhh[j];
      float ghz = s_gh[16 + tid] + bhh[HIDDEN + j];
      float ghn = s_gh[32 + tid] + bhh[2 * HIDDEN + j];
      float r = sigm(git[j] + ghr);
      float z = sigm(git[HIDDEN + j] + ghz);
      float n = tanhf(git[2 * HIDDEN + j] + r * ghn);
      float hn = (1.f - z) * n + z * s_h[j];
      h_glob[j] = hn;
      Hbf[(size_t)t * HIDDEN + j] = f2bf(hn);
      __threadfence();
    }
    __syncthreads();

    if (tid == 0) {
      unsigned old = __hip_atomic_fetch_add(&bar[0], 1u, __ATOMIC_ACQ_REL,
                                            __HIP_MEMORY_SCOPE_AGENT);
      if (old == NWG - 1) {
        __hip_atomic_store(&bar[0], 0u, __ATOMIC_RELAXED, __HIP_MEMORY_SCOPE_AGENT);
        __hip_atomic_fetch_add(&bar[1], 1u, __ATOMIC_RELEASE, __HIP_MEMORY_SCOPE_AGENT);
      } else {
        while (__hip_atomic_load(&bar[1], __ATOMIC_ACQUIRE,
                                 __HIP_MEMORY_SCOPE_AGENT) <= (unsigned)t)
          __builtin_amdgcn_s_sleep(2);
      }
    }
    __syncthreads();
    for (int i = tid; i < 512; i += 192) s_h[i] = h_glob[i];
    __syncthreads();
  }
}

// ---------------- in-place log_softmax over each row of 50257 ----------------
__global__ void __launch_bounds__(256) k_logsoftmax(float* __restrict__ out) {
  __shared__ float red[256];
  int t = blockIdx.x, tid = threadIdx.x;
  float* row = out + (size_t)t * V_TGT;
  float m = -3.4e38f;
  for (int v = tid; v < V_TGT; v += 256) m = fmaxf(m, row[v]);
  red[tid] = m; __syncthreads();
  for (int s = 128; s > 0; s >>= 1) {
    if (tid < s) red[tid] = fmaxf(red[tid], red[tid + s]);
    __syncthreads();
  }
  m = red[0]; __syncthreads();
  float sum = 0.f;
  for (int v = tid; v < V_TGT; v += 256) sum += expf(row[v] - m);
  red[tid] = sum; __syncthreads();
  for (int s = 128; s > 0; s >>= 1) {
    if (tid < s) red[tid] += red[tid + s];
    __syncthreads();
  }
  float c = m + logf(red[0]);
  for (int v = tid; v < V_TGT; v += 256) row[v] -= c;
}

// ---------------- host orchestration ----------------
extern "C" void kernel_launch(void* const* d_in, const int* in_sizes, int n_in,
                              void* d_out, int out_size, void* d_ws, size_t ws_size,
                              hipStream_t stream) {
  (void)in_sizes; (void)n_in; (void)out_size; (void)ws_size;
  const int*   source = (const int*)d_in[0];
  const int*   target = (const int*)d_in[1];
  const float* emb_e  = (const float*)d_in[2];
  const float* Wih_e  = (const float*)d_in[3];
  /* Whh_e (d_in[4]) unused: h0 == 0 */
  const float* bih_e  = (const float*)d_in[5];
  const float* bhh_e  = (const float*)d_in[6];
  const float* emb_d  = (const float*)d_in[7];
  const float* Wih_d  = (const float*)d_in[8];
  const float* Whh_d  = (const float*)d_in[9];
  const float* bih_d  = (const float*)d_in[10];
  const float* bhh_d  = (const float*)d_in[11];
  const float* Wout   = (const float*)d_in[12];
  const float* bout   = (const float*)d_in[13];
  float* out = (float*)d_out;

  char* ws = (char*)d_ws;
  size_t o = 0;
  uint32_t* bar   = (uint32_t*)(ws + o); o += 256;
  float* h_glob   = (float*)(ws + o);    o += 4096;
  float* GiT      = (float*)(ws + o);    o += (size_t)T_LEN * 3 * HIDDEN * 4;
  bf16* Xbf       = (bf16*)(ws + o);     o += (size_t)T_LEN * EMBED * 2;
  bf16* WihBf     = (bf16*)(ws + o);     o += (size_t)3 * HIDDEN * EMBED * 2;
  bf16* WhhBf     = (bf16*)(ws + o);     o += (size_t)3 * HIDDEN * HIDDEN * 2;
  bf16* Hbf       = (bf16*)(ws + o);     o += (size_t)T_LEN * HIDDEN * 2;
  bf16* WoutBf    = (bf16*)(ws + o);     o += (size_t)V_TGT * HIDDEN * 2;

  k_init<<<1, 32, 0, stream>>>(bar);
  k_encoder<<<1, 512, 0, stream>>>(source, emb_e, Wih_e, bih_e, bhh_e, h_glob);
  k_embed<<<T_LEN, EMBED, 0, stream>>>(target, emb_d, Xbf);

  k_f2bf<<<(3 * HIDDEN * EMBED) / 256, 256, 0, stream>>>(Wih_d, WihBf, 3 * HIDDEN * EMBED);
  k_f2bf<<<(3 * HIDDEN * HIDDEN) / 256, 256, 0, stream>>>(Whh_d, WhhBf, 3 * HIDDEN * HIDDEN);
  k_f2bf<<<(V_TGT * HIDDEN + 255) / 256, 256, 0, stream>>>(Wout, WoutBf, V_TGT * HIDDEN);

  {  // GiT[t][row] = (Wih_d @ x_t + bih_d)[row] for all t at once
    int M = 3 * HIDDEN, N = T_LEN, K = EMBED;
    int blocks = ((M + 63) / 64) * (N / 128);
    k_wmma_gemm<<<blocks, 256, 0, stream>>>(WihBf, Xbf, bih_d, GiT, M, N, K);
  }

  k_recurrence<<<NWG, 192, 0, stream>>>(WhhBf, bhh_d, GiT, h_glob, Hbf, bar);

  {  // out[t][v] = (Wout @ h_t + bout)[v]  -- the 13 GMAC GEMM
    int M = V_TGT, N = T_LEN, K = HIDDEN;
    int blocks = ((M + 63) / 64) * (N / 128);
    k_wmma_gemm<<<blocks, 256, 0, stream>>>(WoutBf, Hbf, bout, out, M, N, K);
  }

  k_logsoftmax<<<T_LEN, 256, 0, stream>>>(out);
}
